// HybridGNN_16346645528737
// MI455X (gfx1250) — matrix operand order repaired and checked by reading, not compiled, not run on hardware
//
#include <hip/hip_runtime.h>
#include <hip/hip_bf16.h>

// ---------------------------------------------------------------------------
// Problem constants (match reference)
// ---------------------------------------------------------------------------
#define NN 65536
#define EE 262144
#define GG 2048
#define HH 512
#define HD 1024   /* HEADS * H */
#define DD 256
#define CMB 1280  /* 2H + DESC */

typedef __attribute__((ext_vector_type(16))) _Float16 v16h;
typedef __attribute__((ext_vector_type(8)))  _Float16 v8h;
typedef __attribute__((ext_vector_type(8)))  float    v8f;

// ---------------------------------------------------------------------------
// Device helpers
// ---------------------------------------------------------------------------
__device__ __forceinline__ float gelu_exact(float x) {
    return 0.5f * x * (1.0f + erff(x * 0.70710678118654752440f));
}

// IEEE-ordering float atomic max via int/uint atomics (sign-aware trick)
__device__ __forceinline__ void atomicMaxF(float* addr, float v) {
    if (v >= 0.0f) atomicMax((int*)addr, __float_as_int(v));
    else           atomicMin((unsigned int*)addr, __float_as_uint(v));
}

// A fragment: 16x32 f16, row-major A with leading dim lda.
// Lane l (l<16): row = tileM+l, K = 0..15
// Lane l (>=16): row = tileM+(l-16), K = 16..31
__device__ __forceinline__ v16h load_a_frag(const _Float16* A, int lda,
                                            int row, int k0, int lane) {
    const _Float16* p = A + (size_t)row * lda + k0 + ((lane & 16) ? 16 : 0);
    return *(const v16h*)(p);
}

// B fragment: 32x16 f16 (KxN). B column n == weight row (tileN+n), so a
// column's K run is contiguous in the row-major [Ncols,K] weight matrix.
// Lanes 0..15 hold K=0..15, lanes 16..31 hold K=16..31.
__device__ __forceinline__ v16h load_b_frag(const _Float16* W, int ldb,
                                            int col, int k0, int lane) {
    const _Float16* p = W + (size_t)col * ldb + k0 + ((lane & 16) ? 16 : 0);
    return *(const v16h*)(p);
}

// ---------------------------------------------------------------------------
// Generic WMMA GEMM:  C[M,Nc] = act( A[M,K]*B[Nc,K]^T (+ A2*B2^T) + bias )
// Block = 128 threads = 4 waves stacked in M; each wave computes a 16x64
// output strip (4 accumulators), reusing one A fragment across 4 B tiles
// (4 WMMAs per A load pair).  Requires M%64==0, K%32==0, Nc%64==0.
// act: 0 = identity, 1 = exact GELU.  Ch (optional) gets an f16 shadow copy.
// ---------------------------------------------------------------------------
__global__ void gemm_wmma_f16(const _Float16* __restrict__ A,
                              const _Float16* __restrict__ B,
                              const _Float16* __restrict__ A2,
                              const _Float16* __restrict__ B2,
                              const float* __restrict__ bias,
                              float* __restrict__ C,
                              _Float16* __restrict__ Ch,
                              int M, int K, int Nc, int act) {
    const int lane   = threadIdx.x & 31;
    const int wave   = threadIdx.x >> 5;
    const int tileN0 = blockIdx.x * 64;
    const int tileM  = blockIdx.y * 64 + wave * 16;
    if (tileN0 >= Nc || tileM >= M) return;   // wave-uniform

    const int row  = tileM + (lane & 15);
    const int nsub = lane & 15;

    v8f acc0 = {}, acc1 = {}, acc2 = {}, acc3 = {};
    for (int k0 = 0; k0 < K; k0 += 32) {
        v16h a  = load_a_frag(A, K, row, k0, lane);
        v16h b0 = load_b_frag(B, K, tileN0 +  0 + nsub, k0, lane);
        v16h b1 = load_b_frag(B, K, tileN0 + 16 + nsub, k0, lane);
        v16h b2 = load_b_frag(B, K, tileN0 + 32 + nsub, k0, lane);
        v16h b3 = load_b_frag(B, K, tileN0 + 48 + nsub, k0, lane);
        acc0 = __builtin_amdgcn_wmma_f32_16x16x32_f16(false, a, false, b0, (short)0, acc0, false, false);
        acc1 = __builtin_amdgcn_wmma_f32_16x16x32_f16(false, a, false, b1, (short)0, acc1, false, false);
        acc2 = __builtin_amdgcn_wmma_f32_16x16x32_f16(false, a, false, b2, (short)0, acc2, false, false);
        acc3 = __builtin_amdgcn_wmma_f32_16x16x32_f16(false, a, false, b3, (short)0, acc3, false, false);
    }
    if (A2 != nullptr) {                      // fused second GEMM (SAGE)
        for (int k0 = 0; k0 < K; k0 += 32) {
            v16h a  = load_a_frag(A2, K, row, k0, lane);
            v16h b0 = load_b_frag(B2, K, tileN0 +  0 + nsub, k0, lane);
            v16h b1 = load_b_frag(B2, K, tileN0 + 16 + nsub, k0, lane);
            v16h b2 = load_b_frag(B2, K, tileN0 + 32 + nsub, k0, lane);
            v16h b3 = load_b_frag(B2, K, tileN0 + 48 + nsub, k0, lane);
            acc0 = __builtin_amdgcn_wmma_f32_16x16x32_f16(false, a, false, b0, (short)0, acc0, false, false);
            acc1 = __builtin_amdgcn_wmma_f32_16x16x32_f16(false, a, false, b1, (short)0, acc1, false, false);
            acc2 = __builtin_amdgcn_wmma_f32_16x16x32_f16(false, a, false, b2, (short)0, acc2, false, false);
            acc3 = __builtin_amdgcn_wmma_f32_16x16x32_f16(false, a, false, b3, (short)0, acc3, false, false);
        }
    }

    const int rbase = tileM + ((lane & 16) ? 8 : 0);
    v8f accs[4] = {acc0, acc1, acc2, acc3};
#pragma unroll
    for (int t = 0; t < 4; ++t) {
        const int col = tileN0 + t * 16 + nsub;
        const float bv = (bias != nullptr) ? bias[col] : 0.0f;
#pragma unroll
        for (int r = 0; r < 8; ++r) {
            float v = accs[t][r] + bv;
            if (act == 1) v = gelu_exact(v);
            size_t oi = (size_t)(rbase + r) * Nc + col;
            C[oi] = v;
            if (Ch != nullptr) Ch[oi] = (_Float16)v;
        }
    }
}

// ---------------------------------------------------------------------------
// Utility kernels
// ---------------------------------------------------------------------------
__global__ void fill_f32(float* p, float v, size_t n) {
    size_t i = (size_t)blockIdx.x * blockDim.x + threadIdx.x;
    size_t stride = (size_t)gridDim.x * blockDim.x;
    for (; i < n; i += stride) p[i] = v;
}

__global__ void to_half(const float* __restrict__ in, _Float16* __restrict__ out, size_t n) {
    size_t i = (size_t)blockIdx.x * blockDim.x + threadIdx.x;
    size_t stride = (size_t)gridDim.x * blockDim.x;
    for (; i < n; i += stride) out[i] = (_Float16)in[i];
}

// ---------------------------------------------------------------------------
// Stage 1: GIN
// ---------------------------------------------------------------------------
__global__ void gin_scatter(const int* __restrict__ src, const int* __restrict__ dst,
                            const float* __restrict__ x, float* __restrict__ agg,
                            float* __restrict__ cnt) {
    int e = blockIdx.x * blockDim.x + threadIdx.x;
    if (e >= EE) return;
    int s = src[e], d = dst[e];
#pragma unroll
    for (int c = 0; c < 9; ++c) atomicAdd(&agg[(size_t)d * 9 + c], x[(size_t)s * 9 + c]);
    atomicAdd(&cnt[d], 1.0f);
}

__global__ void gin_linear(const float* __restrict__ x, const float* __restrict__ agg,
                           const float* __restrict__ w, const float* __restrict__ b,
                           float* __restrict__ h, _Float16* __restrict__ hh) {
    int n = blockIdx.x;
    __shared__ float xs[9];
    if (threadIdx.x < 9) xs[threadIdx.x] = x[(size_t)n * 9 + threadIdx.x] + agg[(size_t)n * 9 + threadIdx.x];
    __syncthreads();
    for (int j = threadIdx.x; j < HH; j += blockDim.x) {
        float acc = b[j];
#pragma unroll
        for (int c = 0; c < 9; ++c) acc += xs[c] * w[j * 9 + c];
        float v = gelu_exact(acc);
        size_t oi = (size_t)n * HH + j;
        h[oi] = v;
        hh[oi] = (_Float16)v;
    }
}

// ---------------------------------------------------------------------------
// Stage 2: GAT
// ---------------------------------------------------------------------------
__global__ void gat_attn_coeff(const float* __restrict__ z,
                               const float* __restrict__ asrc,
                               const float* __restrict__ adst,
                               float* __restrict__ a_s, float* __restrict__ a_d) {
    int idx = blockIdx.x * (blockDim.x >> 5) + (threadIdx.x >> 5);
    int lane = threadIdx.x & 31;
    if (idx >= NN * 2) return;
    int n = idx >> 1, hh = idx & 1;
    const float* zp = z + (size_t)n * HD + hh * HH;
    float s = 0.0f, d = 0.0f;
    for (int c = lane; c < HH; c += 32) {
        float zv = zp[c];
        s += zv * asrc[hh * HH + c];
        d += zv * adst[hh * HH + c];
    }
#pragma unroll
    for (int o = 16; o > 0; o >>= 1) {
        s += __shfl_down(s, o, 32);
        d += __shfl_down(d, o, 32);
    }
    if (lane == 0) { a_s[idx] = s; a_d[idx] = d; }
}

__device__ __forceinline__ void edge_ends(const int* src, const int* dst, int e, int& s, int& d) {
    if (e < EE) { s = src[e]; d = dst[e]; }
    else        { s = e - EE; d = e - EE; }        // self loops
}

__global__ void gat_seg_max(const int* __restrict__ src, const int* __restrict__ dst,
                            const float* __restrict__ a_s, const float* __restrict__ a_d,
                            float* __restrict__ m) {
    int e = blockIdx.x * blockDim.x + threadIdx.x;
    if (e >= EE + NN) return;
    int s, d; edge_ends(src, dst, e, s, d);
#pragma unroll
    for (int hh = 0; hh < 2; ++hh) {
        float l = a_s[s * 2 + hh] + a_d[d * 2 + hh];
        l = (l > 0.0f) ? l : 0.2f * l;             // leaky_relu 0.2
        atomicMaxF(&m[d * 2 + hh], l);
    }
}

__global__ void gat_seg_den(const int* __restrict__ src, const int* __restrict__ dst,
                            const float* __restrict__ a_s, const float* __restrict__ a_d,
                            const float* __restrict__ m, float* __restrict__ den) {
    int e = blockIdx.x * blockDim.x + threadIdx.x;
    if (e >= EE + NN) return;
    int s, d; edge_ends(src, dst, e, s, d);
#pragma unroll
    for (int hh = 0; hh < 2; ++hh) {
        float l = a_s[s * 2 + hh] + a_d[d * 2 + hh];
        l = (l > 0.0f) ? l : 0.2f * l;
        atomicAdd(&den[d * 2 + hh], expf(l - m[d * 2 + hh]));
    }
}

// One block per edge: scatter alpha * z[src] into out[dst]  (1024 channels)
__global__ void gat_aggregate(const int* __restrict__ src, const int* __restrict__ dst,
                              const float* __restrict__ a_s, const float* __restrict__ a_d,
                              const float* __restrict__ m, const float* __restrict__ den,
                              const float* __restrict__ z, float* __restrict__ out) {
    int e = blockIdx.x;
    int s, d; edge_ends(src, dst, e, s, d);
    float alpha[2];
#pragma unroll
    for (int hh = 0; hh < 2; ++hh) {
        float l = a_s[s * 2 + hh] + a_d[d * 2 + hh];
        l = (l > 0.0f) ? l : 0.2f * l;
        alpha[hh] = expf(l - m[d * 2 + hh]) / den[d * 2 + hh];
    }
    for (int i = threadIdx.x; i < HD; i += blockDim.x) {
        int hh = i >> 9;
        atomicAdd(&out[(size_t)d * HD + i], alpha[hh] * z[(size_t)s * HD + i]);
    }
}

// h2 = gelu( mean over heads + bias )
__global__ void gat_finish(const float* __restrict__ out, const float* __restrict__ bias,
                           float* __restrict__ h2, _Float16* __restrict__ h2h) {
    size_t i = (size_t)blockIdx.x * blockDim.x + threadIdx.x;
    if (i >= (size_t)NN * HH) return;
    size_t n = i >> 9;
    int c = (int)(i & (HH - 1));
    float v = gelu_exact(0.5f * (out[n * HD + c] + out[n * HD + HH + c]) + bias[c]);
    h2[i] = v;
    h2h[i] = (_Float16)v;
}

// ---------------------------------------------------------------------------
// Stage 3: SAGE
// ---------------------------------------------------------------------------
__global__ void sage_scatter(const int* __restrict__ src, const int* __restrict__ dst,
                             const float* __restrict__ h2, float* __restrict__ sum) {
    int e = blockIdx.x;
    int s = src[e], d = dst[e];
    for (int c = threadIdx.x; c < HH; c += blockDim.x)
        atomicAdd(&sum[(size_t)d * HH + c], h2[(size_t)s * HH + c]);
}

__global__ void sage_mean(const float* __restrict__ sum, const float* __restrict__ cnt,
                          _Float16* __restrict__ mh) {
    size_t i = (size_t)blockIdx.x * blockDim.x + threadIdx.x;
    if (i >= (size_t)NN * HH) return;
    float c = cnt[i >> 9];
    c = (c < 1.0f) ? 1.0f : c;
    mh[i] = (_Float16)(sum[i] / c);
}

// ---------------------------------------------------------------------------
// Stage 4: pooling
// ---------------------------------------------------------------------------
__global__ void pool_nodes(const float* __restrict__ h3, const int* __restrict__ batch,
                           float* __restrict__ pmax, float* __restrict__ psum,
                           float* __restrict__ gcnt) {
    int n = blockIdx.x;
    int g = batch[n];
    if (threadIdx.x == 0) atomicAdd(&gcnt[g], 1.0f);
    for (int c = threadIdx.x; c < HH; c += blockDim.x) {
        float v = h3[(size_t)n * HH + c];
        atomicMaxF(&pmax[(size_t)g * HH + c], v);
        atomicAdd(&psum[(size_t)g * HH + c], v);
    }
}

__global__ void build_combined(const float* __restrict__ pmax, const float* __restrict__ psum,
                               const float* __restrict__ gcnt, const float* __restrict__ desc,
                               _Float16* __restrict__ ch) {
    size_t i = (size_t)blockIdx.x * blockDim.x + threadIdx.x;
    if (i >= (size_t)GG * CMB) return;
    size_t g = i / CMB;
    int c = (int)(i % CMB);
    float v;
    if (c < HH)            v = pmax[g * HH + c];
    else if (c < 2 * HH)   v = psum[g * HH + (c - HH)] / gcnt[g];
    else                   v = desc[g * DD + (c - 2 * HH)];
    ch[i] = (_Float16)v;
}

// ---------------------------------------------------------------------------
// Stage 5 tail: out = z1 @ a2_w^T + a2_b  (one wave per graph)
// ---------------------------------------------------------------------------
__global__ void final_head(const float* __restrict__ z1, const float* __restrict__ w,
                           const float* __restrict__ b, float* __restrict__ out) {
    int g = blockIdx.x * (blockDim.x >> 5) + (threadIdx.x >> 5);
    int lane = threadIdx.x & 31;
    if (g >= GG) return;
    float acc = 0.0f;
    for (int c = lane; c < 256; c += 32) acc += z1[(size_t)g * 256 + c] * w[c];
#pragma unroll
    for (int o = 16; o > 0; o >>= 1) acc += __shfl_down(acc, o, 32);
    if (lane == 0) out[g] = acc + b[0];
}

// ---------------------------------------------------------------------------
// Workspace layout (all offsets keep >=32B alignment)
// ---------------------------------------------------------------------------
#define SZ(x) ((size_t)(x))
static const size_t OFF_AGG9  = 0;                                   // N*9 f32
static const size_t OFF_CNT   = OFF_AGG9  + SZ(NN) * 9 * 4;          // N f32
static const size_t OFF_H1F   = OFF_CNT   + SZ(NN) * 4;              // N*H f32  (reused: h2)
static const size_t OFF_H1H   = OFF_H1F   + SZ(NN) * HH * 4;         // N*H f16  (reused: h2h)
static const size_t OFF_Z     = OFF_H1H   + SZ(NN) * HH * 2;         // N*HD f32 (reused: h3)
static const size_t OFF_AS    = OFF_Z     + SZ(NN) * HD * 4;         // N*2 f32
static const size_t OFF_AD    = OFF_AS    + SZ(NN) * 2 * 4;
static const size_t OFF_M     = OFF_AD    + SZ(NN) * 2 * 4;
static const size_t OFF_DEN   = OFF_M     + SZ(NN) * 2 * 4;
static const size_t OFF_GOUT  = OFF_DEN   + SZ(NN) * 2 * 4;          // N*HD f32 (reused: mean_sum + mean_h)
static const size_t OFF_MEANH = OFF_GOUT  + SZ(NN) * HH * 4;         // f16 alias inside GOUT
static const size_t OFF_WGAT  = OFF_GOUT  + SZ(NN) * HD * 4;         // f16 weights
static const size_t OFF_WL    = OFF_WGAT  + SZ(HD) * HH * 2;
static const size_t OFF_WR    = OFF_WL    + SZ(HH) * HH * 2;
static const size_t OFF_WFC   = OFF_WR    + SZ(HH) * HH * 2;
static const size_t OFF_WA1   = OFF_WFC   + SZ(512) * CMB * 2;
static const size_t OFF_PMAX  = OFF_WA1   + SZ(256) * 512 * 2;
static const size_t OFF_PSUM  = OFF_PMAX  + SZ(GG) * HH * 4;
static const size_t OFF_GCNT  = OFF_PSUM  + SZ(GG) * HH * 4;
static const size_t OFF_COMBH = OFF_GCNT  + SZ(GG) * 4;
static const size_t OFF_SFF   = OFF_COMBH + SZ(GG) * CMB * 2;
static const size_t OFF_SFH   = OFF_SFF   + SZ(GG) * HH * 4;
static const size_t OFF_Z1    = OFF_SFH   + SZ(GG) * HH * 2;

// ---------------------------------------------------------------------------
// Host launcher
// ---------------------------------------------------------------------------
extern "C" void kernel_launch(void* const* d_in, const int* in_sizes, int n_in,
                              void* d_out, int out_size, void* d_ws, size_t ws_size,
                              hipStream_t stream) {
    (void)in_sizes; (void)n_in; (void)out_size; (void)ws_size;
    const float* x        = (const float*)d_in[0];
    const int*   eidx     = (const int*)d_in[1];     // [2,E]
    const int*   batch    = (const int*)d_in[2];
    const float* desc     = (const float*)d_in[3];
    const float* gin_w    = (const float*)d_in[4];
    const float* gin_b    = (const float*)d_in[5];
    const float* gat_w    = (const float*)d_in[6];
    const float* gat_asrc = (const float*)d_in[7];
    const float* gat_adst = (const float*)d_in[8];
    const float* gat_bias = (const float*)d_in[9];
    const float* sage_wl  = (const float*)d_in[10];
    const float* sage_bl  = (const float*)d_in[11];
    const float* sage_wr  = (const float*)d_in[12];
    const float* fc_w     = (const float*)d_in[13];
    const float* fc_b     = (const float*)d_in[14];
    const float* a1_w     = (const float*)d_in[15];
    const float* a1_b     = (const float*)d_in[16];
    const float* a2_w     = (const float*)d_in[17];
    const float* a2_b     = (const float*)d_in[18];
    float* out = (float*)d_out;

    const int* src = eidx;
    const int* dst = eidx + EE;

    char* ws = (char*)d_ws;
    float*    agg9   = (float*)(ws + OFF_AGG9);
    float*    cnt    = (float*)(ws + OFF_CNT);
    float*    h1f    = (float*)(ws + OFF_H1F);     // later h2
    _Float16* h1h    = (_Float16*)(ws + OFF_H1H);  // later h2h
    float*    z      = (float*)(ws + OFF_Z);       // later h3
    float*    a_s    = (float*)(ws + OFF_AS);
    float*    a_d    = (float*)(ws + OFF_AD);
    float*    mbuf   = (float*)(ws + OFF_M);
    float*    den    = (float*)(ws + OFF_DEN);
    float*    gout   = (float*)(ws + OFF_GOUT);    // later mean_sum
    _Float16* meanh  = (_Float16*)(ws + OFF_MEANH);
    _Float16* wgat_h = (_Float16*)(ws + OFF_WGAT);
    _Float16* wl_h   = (_Float16*)(ws + OFF_WL);
    _Float16* wr_h   = (_Float16*)(ws + OFF_WR);
    _Float16* wfc_h  = (_Float16*)(ws + OFF_WFC);
    _Float16* wa1_h  = (_Float16*)(ws + OFF_WA1);
    float*    pmax   = (float*)(ws + OFF_PMAX);
    float*    psum   = (float*)(ws + OFF_PSUM);
    float*    gcnt   = (float*)(ws + OFF_GCNT);
    _Float16* combh  = (_Float16*)(ws + OFF_COMBH);
    float*    sff    = (float*)(ws + OFF_SFF);
    _Float16* sfh    = (_Float16*)(ws + OFF_SFH);
    float*    z1     = (float*)(ws + OFF_Z1);

    const float NEG_INF = -__builtin_huge_valf();
    auto blocks = [](size_t n, int t) { return (unsigned)((n + t - 1) / t); };

    // --- init accumulators ---
    fill_f32<<<blocks(SZ(NN) * 9, 256), 256, 0, stream>>>(agg9, 0.0f, SZ(NN) * 9);
    fill_f32<<<blocks(NN, 256), 256, 0, stream>>>(cnt, 0.0f, NN);
    fill_f32<<<blocks(SZ(NN) * 2, 256), 256, 0, stream>>>(mbuf, NEG_INF, SZ(NN) * 2);
    fill_f32<<<blocks(SZ(NN) * 2, 256), 256, 0, stream>>>(den, 0.0f, SZ(NN) * 2);
    fill_f32<<<blocks(SZ(NN) * HD, 256), 256, 0, stream>>>(gout, 0.0f, SZ(NN) * HD);
    fill_f32<<<blocks(SZ(GG) * HH, 256), 256, 0, stream>>>(pmax, NEG_INF, SZ(GG) * HH);
    fill_f32<<<blocks(SZ(GG) * HH, 256), 256, 0, stream>>>(psum, 0.0f, SZ(GG) * HH);
    fill_f32<<<blocks(GG, 256), 256, 0, stream>>>(gcnt, 0.0f, GG);

    // --- weights -> f16 ---
    to_half<<<blocks(SZ(HD) * HH, 256), 256, 0, stream>>>(gat_w, wgat_h, SZ(HD) * HH);
    to_half<<<blocks(SZ(HH) * HH, 256), 256, 0, stream>>>(sage_wl, wl_h, SZ(HH) * HH);
    to_half<<<blocks(SZ(HH) * HH, 256), 256, 0, stream>>>(sage_wr, wr_h, SZ(HH) * HH);
    to_half<<<blocks(SZ(512) * CMB, 256), 256, 0, stream>>>(fc_w, wfc_h, SZ(512) * CMB);
    to_half<<<blocks(SZ(256) * 512, 256), 256, 0, stream>>>(a1_w, wa1_h, SZ(256) * 512);

    // --- Stage 1: GIN ---
    gin_scatter<<<blocks(EE, 256), 256, 0, stream>>>(src, dst, x, agg9, cnt);
    gin_linear<<<NN, 256, 0, stream>>>(x, agg9, gin_w, gin_b, h1f, h1h);

    // --- Stage 2: GAT ---
    // z = h1 @ gat_w^T : [N, 1024]
    {
        dim3 g(HD / 64, NN / 64);
        gemm_wmma_f16<<<g, 128, 0, stream>>>(h1h, wgat_h, nullptr, nullptr,
                                             nullptr, z, nullptr, NN, HH, HD, 0);
    }
    gat_attn_coeff<<<blocks(SZ(NN) * 2 * 32, 256), 256, 0, stream>>>(z, gat_asrc, gat_adst, a_s, a_d);
    gat_seg_max<<<blocks(EE + NN, 256), 256, 0, stream>>>(src, dst, a_s, a_d, mbuf);
    gat_seg_den<<<blocks(EE + NN, 256), 256, 0, stream>>>(src, dst, a_s, a_d, mbuf, den);
    gat_aggregate<<<EE + NN, 256, 0, stream>>>(src, dst, a_s, a_d, mbuf, den, z, gout);
    gat_finish<<<blocks(SZ(NN) * HH, 256), 256, 0, stream>>>(gout, gat_bias, h1f, h1h); // h1 buffers now hold h2

    // --- Stage 3: SAGE ---
    fill_f32<<<blocks(SZ(NN) * HH, 256), 256, 0, stream>>>(gout, 0.0f, SZ(NN) * HH);    // mean_sum aliases gout
    sage_scatter<<<EE, 256, 0, stream>>>(src, dst, h1f, gout);
    sage_mean<<<blocks(SZ(NN) * HH, 256), 256, 0, stream>>>(gout, cnt, meanh);
    // h3 = gelu(mean @ wl^T + h2 @ wr^T + bl), written into z buffer
    {
        dim3 g(HH / 64, NN / 64);
        gemm_wmma_f16<<<g, 128, 0, stream>>>(meanh, wl_h, h1h, wr_h,
                                             sage_bl, z, nullptr, NN, HH, HH, 1);
    }

    // --- Stage 4: pooling + combined ---
    pool_nodes<<<NN, 256, 0, stream>>>(z, batch, pmax, psum, gcnt);
    build_combined<<<blocks(SZ(GG) * CMB, 256), 256, 0, stream>>>(pmax, psum, gcnt, desc, combh);

    // --- Stage 5: dense head ---
    {
        dim3 g(512 / 64, GG / 64);
        gemm_wmma_f16<<<g, 128, 0, stream>>>(combh, wfc_h, nullptr, nullptr,
                                             fc_b, sff, sfh, GG, CMB, 512, 1);
    }
    {
        dim3 g(256 / 64, GG / 64);
        gemm_wmma_f16<<<g, 128, 0, stream>>>(sfh, wa1_h, nullptr, nullptr,
                                             a1_b, z1, nullptr, GG, HH, 256, 1);
    }
    final_head<<<blocks(SZ(GG) * 32, 256), 256, 0, stream>>>(z1, a2_w, a2_b, out);
}
